// DCBR_31241592111771
// MI455X (gfx1250) — compile-verified, matrix operand order checked by reading, not run on hardware
//
#include <hip/hip_runtime.h>
#include <hip/hip_bf16.h>

#define DIM 64

constexpr int   Unum = 50000, Bnum = 20000, Inum = 100000, Lnum = 2;
constexpr int   BATCH = 2048;
constexpr float TAU = 0.25f, OMEGA = 0.5f, G1 = 0.5f, G2 = 0.5f;
constexpr float LAM1 = 0.04f, LAM2 = 1e-5f;
constexpr float XI = 1.0f / 3.0f;
constexpr float UPS = 0.1f;

typedef __attribute__((ext_vector_type(16))) _Float16 v16h;
typedef __attribute__((ext_vector_type(8)))  float    v8f;

// ---------------- sparse phase ----------------

__global__ void init_prop_kernel(const float* __restrict__ A, const float* __restrict__ Bf,
                                 int nA, int nB, float* __restrict__ cur, float* __restrict__ acc) {
    long long idx = (long long)blockIdx.x * blockDim.x + threadIdx.x;
    long long total = (long long)(nA + nB) * DIM;
    if (idx >= total) return;
    long long aElems = (long long)nA * DIM;
    float v = (idx < aElems) ? A[idx] : Bf[idx - aElems];
    cur[idx] = v;
    acc[idx] = XI * v;
}

// 16 threads per edge, 4 dims each (b128 gather + 4 scatter-atomics)
__global__ void spmm_edge_kernel(const int* __restrict__ rows, const int* __restrict__ cols,
                                 const float* __restrict__ vals, const float* __restrict__ x,
                                 float* __restrict__ y, long long nnz) {
    long long idx = (long long)blockIdx.x * blockDim.x + threadIdx.x;
    if (idx >= nnz * 16) return;
    long long e = idx >> 4;
    int q = (int)(idx & 15) << 2;
    float v = vals[e];
    const float4 xv = *(const float4*)(x + (long long)cols[e] * DIM + q);
    float* yp = y + (long long)rows[e] * DIM + q;
    atomicAdd(yp + 0, v * xv.x);
    atomicAdd(yp + 1, v * xv.y);
    atomicAdd(yp + 2, v * xv.z);
    atomicAdd(yp + 3, v * xv.w);
}

// one block (64 threads) per row: feats = sp + sign(sp)*l2norm(noise)*ups;
// cur = feats; acc += XI * l2norm(feats)
__global__ void perturb_accum_kernel(const float* __restrict__ sp, const float* __restrict__ noise,
                                     float ups, float* __restrict__ cur, float* __restrict__ acc) {
    __shared__ float red[DIM];
    int r = blockIdx.x, t = threadIdx.x;
    long long base = (long long)r * DIM + t;
    float nv = noise[base];
    red[t] = nv * nv; __syncthreads();
    for (int s = 32; s > 0; s >>= 1) { if (t < s) red[t] += red[t + s]; __syncthreads(); }
    float nn = sqrtf(red[0]); __syncthreads();
    float ln = nv / fmaxf(nn, 1e-12f);
    float v = sp[base];
    float sg = (v > 0.f) ? 1.f : ((v < 0.f) ? -1.f : 0.f);
    float f = v + sg * ln * ups;
    cur[base] = f;
    red[t] = f * f; __syncthreads();
    for (int s = 32; s > 0; s >>= 1) { if (t < s) red[t] += red[t + s]; __syncthreads(); }
    float fn = sqrtf(red[0]);
    acc[base] += XI * (f / fmaxf(fn, 1e-12f));
}

// in-place: out = out + sign(out)*l2norm(noise)*ups
__global__ void agg_perturb_kernel(const float* __restrict__ noise, float ups, float* __restrict__ out) {
    __shared__ float red[DIM];
    int r = blockIdx.x, t = threadIdx.x;
    long long base = (long long)r * DIM + t;
    float nv = noise[base];
    red[t] = nv * nv; __syncthreads();
    for (int s = 32; s > 0; s >>= 1) { if (t < s) red[t] += red[t + s]; __syncthreads(); }
    float nn = sqrtf(red[0]);
    float ln = nv / fmaxf(nn, 1e-12f);
    float v = out[base];
    float sg = (v > 0.f) ? 1.f : ((v < 0.f) ? -1.f : 0.f);
    out[base] = v + sg * ln * ups;
}

// ---------------- BPR ----------------

__global__ void sumsq_kernel(const float* __restrict__ p, long long n, float* __restrict__ acc) {
    __shared__ float red[256];
    int t = threadIdx.x;
    float s = 0.f;
    for (long long i = (long long)blockIdx.x * blockDim.x + t; i < n; i += (long long)gridDim.x * blockDim.x)
        s += p[i] * p[i];
    red[t] = s; __syncthreads();
    for (int st = 128; st > 0; st >>= 1) { if (t < st) red[t] += red[t + st]; __syncthreads(); }
    if (t == 0) atomicAdd(acc, red[0]);
}

__global__ void bpr_kernel(const float* __restrict__ UIu, const float* __restrict__ BIu,
                           const float* __restrict__ UIb, const float* __restrict__ BIb,
                           const int* __restrict__ users, const int* __restrict__ bundles,
                           float* __restrict__ scal) {
    __shared__ float rp[DIM], rn[DIM];
    int i = blockIdx.x, t = threadIdx.x;
    int u  = users[i];
    int b0 = bundles[2 * i + 0];
    int b1 = bundles[2 * i + 1];
    float ur  = OMEGA * UIu[(long long)u  * DIM + t] + (1.f - OMEGA) * BIu[(long long)u  * DIM + t];
    float br0 = OMEGA * UIb[(long long)b0 * DIM + t] + (1.f - OMEGA) * BIb[(long long)b0 * DIM + t];
    float br1 = OMEGA * UIb[(long long)b1 * DIM + t] + (1.f - OMEGA) * BIb[(long long)b1 * DIM + t];
    rp[t] = ur * br0; rn[t] = ur * br1; __syncthreads();
    for (int s = 32; s > 0; s >>= 1) { if (t < s) { rp[t] += rp[t + s]; rn[t] += rn[t + s]; } __syncthreads(); }
    if (t == 0) {
        float x = rp[0] - rn[0];
        float ls = (x > 0.f) ? -log1pf(__expf(-x)) : (x - log1pf(__expf(x)));  // log_sigmoid
        atomicAdd(&scal[0], -ls);
    }
}

// ---------------- contrastive phase ----------------

__global__ void gather_norm_kernel(const float* __restrict__ src, const int* __restrict__ idx, int stride,
                                   float* __restrict__ o32, _Float16* __restrict__ o16) {
    __shared__ float red[DIM];
    int i = blockIdx.x, t = threadIdx.x;
    int row = idx[i * stride];
    float v = src[(long long)row * DIM + t];
    red[t] = v * v; __syncthreads();
    for (int s = 32; s > 0; s >>= 1) { if (t < s) red[t] += red[t + s]; __syncthreads(); }
    float w = v / fmaxf(sqrtf(red[0]), 1e-12f);
    o32[i * DIM + t] = w;
    o16[i * DIM + t] = (_Float16)w;
}

__global__ void diag_dot_kernel(const float* __restrict__ X, const float* __restrict__ Y,
                                float* __restrict__ dg) {
    __shared__ float red[DIM];
    int i = blockIdx.x, t = threadIdx.x;
    red[t] = X[i * DIM + t] * Y[i * DIM + t]; __syncthreads();
    for (int s = 32; s > 0; s >>= 1) { if (t < s) red[t] += red[t + s]; __syncthreads(); }
    if (t == 0) dg[i] = red[0];
}

// A (M x K row-major) and B = Y^T (K x N with N = Y-row) share the identical
// per-lane 16-bit fragment layout: lane -> row (m = lane&15, half = lane>>4),
// element j<8 -> K = 8*half + j, j>=8 -> K = 16 + 8*half + (j-8). Two 16-byte
// contiguous chunks per lane per k-step.
__device__ inline v16h load_frag16(const _Float16* __restrict__ M, int r0, int lane, int kbase) {
    int m = lane & 15, h = lane >> 4;
    const _Float16* p = M + (long long)(r0 + m) * DIM + kbase + 8 * h;
    v16h a;
#pragma unroll
    for (int j = 0; j < 8; ++j) { a[j] = p[j]; a[j + 8] = p[16 + j]; }
    return a;
}

__device__ inline const _Float16* selg(int i,
        const _Float16* g0, const _Float16* g1, const _Float16* g2,
        const _Float16* g3, const _Float16* g4, const _Float16* g5) {
    switch (i) {
        case 0: return g0; case 1: return g1; case 2: return g2;
        case 3: return g3; case 4: return g4; default: return g5;
    }
}

// XOR-lane reduction across the 16 lanes of each half using ds_swizzle
// (group-of-32 mode: imm = xor_mask<<10 | and_mask 0x1F)
__device__ inline float xor16_sum(float e) {
    e += __int_as_float(__builtin_amdgcn_ds_swizzle(__float_as_int(e), (1 << 10) | 0x1F));
    e += __int_as_float(__builtin_amdgcn_ds_swizzle(__float_as_int(e), (2 << 10) | 0x1F));
    e += __int_as_float(__builtin_amdgcn_ds_swizzle(__float_as_int(e), (4 << 10) | 0x1F));
    e += __int_as_float(__builtin_amdgcn_ds_swizzle(__float_as_int(e), (8 << 10) | 0x1F));
    return e;
}

// All 12 cl_loss GEMMs in one kernel. grid = (rowTiles=128, colBlocks=2, pairs=12),
// block = 128 (4 waves). Each wave owns one row tile and a chunk of 16 column
// tiles: A fragments stay in registers (32 WMMAs per wave), exp(S/tau) row sums
// accumulate in registers, single cross-lane reduce + atomic commit at the end.
__global__ __launch_bounds__(128) void wmma_ttl_kernel(
        const _Float16* __restrict__ g0, const _Float16* __restrict__ g1,
        const _Float16* __restrict__ g2, const _Float16* __restrict__ g3,
        const _Float16* __restrict__ g4, const _Float16* __restrict__ g5,
        unsigned long long px, unsigned long long py, float* __restrict__ ttl) {
    int pair = blockIdx.z;
    const _Float16* X = selg((int)((px >> (3 * pair)) & 7), g0, g1, g2, g3, g4, g5);
    const _Float16* Y = selg((int)((py >> (3 * pair)) & 7), g0, g1, g2, g3, g4, g5);

    int wave = threadIdx.x >> 5;
    int lane = threadIdx.x & 31;
    int r0 = blockIdx.x * 16;
    int c0base = (blockIdx.y * 4 + wave) * 256;   // 16 column tiles of 16

    v16h a0 = load_frag16(X, r0, lane, 0);
    v16h a1 = load_frag16(X, r0, lane, 32);

    float s[8];
#pragma unroll
    for (int r = 0; r < 8; ++r) s[r] = 0.f;

    for (int tile = 0; tile < 16; ++tile) {
        int c0 = c0base + tile * 16;
        v16h b0 = load_frag16(Y, c0, lane, 0);
        v16h b1 = load_frag16(Y, c0, lane, 32);
        v8f c = {};
        c = __builtin_amdgcn_wmma_f32_16x16x32_f16(false, a0, false, b0, (short)0, c, false, false);
        c = __builtin_amdgcn_wmma_f32_16x16x32_f16(false, a1, false, b1, (short)0, c, false, false);
#pragma unroll
        for (int r = 0; r < 8; ++r) s[r] += __expf(c[r] * (1.0f / TAU));
    }

    // c[r] at lane -> S[M = r + 8*(lane>>4)][N = lane&15]; sum over N per row
    int m = lane & 15, h = lane >> 4;
    float* trow = ttl + pair * BATCH + r0;
#pragma unroll
    for (int r = 0; r < 8; ++r) {
        float e = xor16_sum(s[r]);
        if (m == 0) atomicAdd(&trow[r + 8 * h], e);
    }
}

__global__ void finalize_kernel(const float* __restrict__ diag, const float* __restrict__ ttl,
                                const float* __restrict__ scal, float* __restrict__ out) {
    __shared__ float ri[256], ra[256];
    int t = threadIdx.x;
    float si = 0.f, sa = 0.f;
    for (int k = t; k < 12 * BATCH; k += 256) {
        float term = -(diag[k] * (1.0f / TAU) - logf(fmaxf(ttl[k], 1e-30f)));
        if (k < 6 * BATCH) si += term; else sa += term;
    }
    ri[t] = si; ra[t] = sa; __syncthreads();
    for (int s = 128; s > 0; s >>= 1) { if (t < s) { ri[t] += ri[t + s]; ra[t] += ra[t + s]; } __syncthreads(); }
    if (t == 0) {
        out[0] = scal[0] / (float)BATCH + LAM2 * scal[1];                     // bpr
        out[1] = LAM1 * (G1 * ri[0] + G2 * ra[0]) / (float)BATCH;             // cl
    }
}

// ---------------- host launcher ----------------

extern "C" void kernel_launch(void* const* d_in, const int* in_sizes, int n_in,
                              void* d_out, int out_size, void* d_ws, size_t ws_size,
                              hipStream_t stream) {
    const float* users_f   = (const float*)d_in[0];
    const float* bundles_f = (const float*)d_in[1];
    const float* items_f   = (const float*)d_in[2];
    const int*   ub_rows = (const int*)d_in[3];  const int* ub_cols = (const int*)d_in[4];
    const float* ub_vals = (const float*)d_in[5];
    const int*   ui_rows = (const int*)d_in[6];  const int* ui_cols = (const int*)d_in[7];
    const float* ui_vals = (const float*)d_in[8];
    const int*   bi_rows = (const int*)d_in[9];  const int* bi_cols = (const int*)d_in[10];
    const float* bi_vals = (const float*)d_in[11];
    const int*   biagg_rows = (const int*)d_in[12]; const int* biagg_cols = (const int*)d_in[13];
    const float* biagg_vals = (const float*)d_in[14];
    const int*   uiagg_rows = (const int*)d_in[15]; const int* uiagg_cols = (const int*)d_in[16];
    const float* uiagg_vals = (const float*)d_in[17];
    const float* noise_UB     = (const float*)d_in[18];
    const float* noise_UI     = (const float*)d_in[19];
    const float* noise_BI     = (const float*)d_in[20];
    const float* noise_agg_BI = (const float*)d_in[21];
    const float* noise_agg_UI = (const float*)d_in[22];
    const int*   users   = (const int*)d_in[23];
    const int*   bundles = (const int*)d_in[24];

    long long nnz_ub = in_sizes[3], nnz_ui = in_sizes[6], nnz_bi = in_sizes[9];
    long long nnz_biagg = in_sizes[12], nnz_uiagg = in_sizes[15];

    // workspace carve-up
    char* ws = (char*)d_ws;
    size_t off = 0;
    auto alloc = [&](size_t bytes) -> void* {
        off = (off + 255) & ~(size_t)255;
        void* p = ws + off;
        off += bytes;
        return p;
    };
    float* UB_out = (float*)alloc((size_t)(Unum + Bnum) * DIM * 4);
    float* UI_out = (float*)alloc((size_t)(Unum + Inum) * DIM * 4);
    float* BI_out = (float*)alloc((size_t)(Bnum + Inum) * DIM * 4);
    float* UIb    = (float*)alloc((size_t)Bnum * DIM * 4);   // graph_aggregate -> UI_b
    float* BIu    = (float*)alloc((size_t)Unum * DIM * 4);   // graph_aggregate -> BI_u
    float* cur    = (float*)alloc((size_t)(Unum + Inum) * DIM * 4);
    float* tmp    = (float*)alloc((size_t)(Unum + Inum) * DIM * 4);
    float*    g32[6];
    _Float16* g16[6];
    for (int i = 0; i < 6; ++i) g32[i] = (float*)alloc((size_t)BATCH * DIM * 4);
    for (int i = 0; i < 6; ++i) g16[i] = (_Float16*)alloc((size_t)BATCH * DIM * 2);
    float* diag = (float*)alloc((size_t)12 * BATCH * 4);
    float* ttl  = (float*)alloc((size_t)12 * BATCH * 4);
    float* scal = (float*)alloc(256);   // [0]=bpr partial, [1]=reg

    hipMemsetAsync(scal, 0, 256, stream);
    hipMemsetAsync(ttl, 0, (size_t)12 * BATCH * 4, stream);

    auto propagate = [&](const float* A, const float* Bf, int nA, int nB,
                         const int* rows, const int* cols, const float* vals, long long nnz,
                         const float* noise, float ups, float* out) {
        int n = nA + nB;
        long long tot = (long long)n * DIM;
        init_prop_kernel<<<(unsigned)((tot + 255) / 256), 256, 0, stream>>>(A, Bf, nA, nB, cur, out);
        for (int l = 0; l < Lnum; ++l) {
            hipMemsetAsync(tmp, 0, (size_t)tot * 4, stream);
            long long work = nnz * 16;
            spmm_edge_kernel<<<(unsigned)((work + 255) / 256), 256, 0, stream>>>(rows, cols, vals, cur, tmp, nnz);
            perturb_accum_kernel<<<n, DIM, 0, stream>>>(tmp, noise + (long long)l * n * DIM, ups, cur, out);
        }
    };

    propagate(users_f,   bundles_f, Unum, Bnum, ub_rows, ub_cols, ub_vals, nnz_ub, noise_UB, UPS, UB_out);
    propagate(users_f,   items_f,   Unum, Inum, ui_rows, ui_cols, ui_vals, nnz_ui, noise_UI, UPS, UI_out);
    propagate(bundles_f, items_f,   Bnum, Inum, bi_rows, bi_cols, bi_vals, nnz_bi, noise_BI, UPS, BI_out);

    // UI_b = aggregate over bi-agg edges of UI items slice
    hipMemsetAsync(UIb, 0, (size_t)Bnum * DIM * 4, stream);
    spmm_edge_kernel<<<(unsigned)((nnz_biagg * 16 + 255) / 256), 256, 0, stream>>>(
        biagg_rows, biagg_cols, biagg_vals, UI_out + (long long)Unum * DIM, UIb, nnz_biagg);
    agg_perturb_kernel<<<Bnum, DIM, 0, stream>>>(noise_agg_BI, UPS, UIb);

    // BI_u = aggregate over ui-agg edges of BI items slice
    hipMemsetAsync(BIu, 0, (size_t)Unum * DIM * 4, stream);
    spmm_edge_kernel<<<(unsigned)((nnz_uiagg * 16 + 255) / 256), 256, 0, stream>>>(
        uiagg_rows, uiagg_cols, uiagg_vals, BI_out + (long long)Bnum * DIM, BIu, nnz_uiagg);
    agg_perturb_kernel<<<Unum, DIM, 0, stream>>>(noise_agg_UI, UPS, BIu);

    // regularizer + BPR
    sumsq_kernel<<<512, 256, 0, stream>>>(users_f,   (long long)Unum * DIM, &scal[1]);
    sumsq_kernel<<<512, 256, 0, stream>>>(bundles_f, (long long)Bnum * DIM, &scal[1]);
    sumsq_kernel<<<512, 256, 0, stream>>>(items_f,   (long long)Inum * DIM, &scal[1]);
    bpr_kernel<<<BATCH, DIM, 0, stream>>>(UI_out, BIu, UIb, BI_out, users, bundles, &scal[0]);

    // gather + normalize the six 2048x64 matrices
    // 0=UB_u[u0] 1=UI_u[u0] 2=BI_u[u0] 3=UB_b[b0] 4=UI_b[b0] 5=BI_b[b0]
    gather_norm_kernel<<<BATCH, DIM, 0, stream>>>(UB_out,                         users,   1, g32[0], g16[0]);
    gather_norm_kernel<<<BATCH, DIM, 0, stream>>>(UI_out,                         users,   1, g32[1], g16[1]);
    gather_norm_kernel<<<BATCH, DIM, 0, stream>>>(BIu,                            users,   1, g32[2], g16[2]);
    gather_norm_kernel<<<BATCH, DIM, 0, stream>>>(UB_out + (long long)Unum * DIM, bundles, 2, g32[3], g16[3]);
    gather_norm_kernel<<<BATCH, DIM, 0, stream>>>(UIb,                            bundles, 2, g32[4], g16[4]);
    gather_norm_kernel<<<BATCH, DIM, 0, stream>>>(BI_out,                         bundles, 2, g32[5], g16[5]);

    // 12 cl_loss pairs: first 6 inter, last 6 intra
    const int pairX[12] = {0, 0, 1, 3, 3, 4, 0, 1, 2, 3, 4, 5};
    const int pairY[12] = {1, 2, 2, 4, 5, 5, 0, 1, 2, 3, 4, 5};
    unsigned long long px = 0, py = 0;
    for (int p = 0; p < 12; ++p) {
        px |= (unsigned long long)pairX[p] << (3 * p);
        py |= (unsigned long long)pairY[p] << (3 * p);
        diag_dot_kernel<<<BATCH, DIM, 0, stream>>>(g32[pairX[p]], g32[pairY[p]], diag + p * BATCH);
    }
    dim3 gtile(BATCH / 16, 2, 12);
    wmma_ttl_kernel<<<gtile, 128, 0, stream>>>(g16[0], g16[1], g16[2], g16[3], g16[4], g16[5],
                                               px, py, ttl);

    finalize_kernel<<<1, 256, 0, stream>>>(diag, ttl, scal, (float*)d_out);
}
